// MultiHeadSelfAttention_31714038513757
// MI455X (gfx1250) — matrix operand order repaired and checked by reading, not compiled, not run on hardware
//
#include <hip/hip_runtime.h>

#define S_LEN   4096
#define DMODEL  1024
#define NHEADS  16
#define HDIM    64

typedef __attribute__((ext_vector_type(16))) __bf16 v16bf;
typedef __attribute__((ext_vector_type(8)))  __bf16 v8bf;
typedef __attribute__((ext_vector_type(8)))  float  v8f;

static __device__ __forceinline__ __bf16 f2bf(float f) {
  union { float f; unsigned u; } v; v.f = f;
  unsigned r = v.u + 0x7FFFu + ((v.u >> 16) & 1u);   // round-to-nearest-even
  unsigned short h = (unsigned short)(r >> 16);
  return __builtin_bit_cast(__bf16, h);
}
static __device__ __forceinline__ float bf2f(__bf16 b) {
  unsigned short h = __builtin_bit_cast(unsigned short, b);
  union { unsigned u; float f; } v; v.u = ((unsigned)h) << 16;
  return v.f;
}
static __device__ __forceinline__ v8f wmma_bf16(v16bf a, v16bf b, v8f c) {
  // D = A(16x32 bf16) * B(32x16 bf16) + C(16x16 f32)
  return __builtin_amdgcn_wmma_f32_16x16x32_bf16(false, a, false, b, (short)0, c,
                                                 false, false);
}
// A-fragment: two 16-byte K-chunks (K=half*8..+8 and K=16+half*8..+8)
static __device__ __forceinline__ v16bf load_frag2(const __bf16* p, int elems_to_hi) {
  v16bf r;
  ((v8bf*)&r)[0] = *(const v8bf*)p;
  ((v8bf*)&r)[1] = *(const v8bf*)(p + elems_to_hi);
  return r;
}

// CDNA5 async global->LDS copy (16B per lane), tracked by ASYNCcnt.
// Flat LDS pointers carry the LDS byte offset in addr[31:0] (ISA 10.2 aperture
// rules), so truncation yields a valid DS address for the VDST operand.
static __device__ __forceinline__ void async_copy_b128(const void* gaddr, void* lds) {
  unsigned l = (unsigned)(uintptr_t)lds;
  asm volatile("global_load_async_to_lds_b128 %0, %1, off"
               :: "v"(l), "v"(gaddr) : "memory");
}
static __device__ __forceinline__ void wait_async_le4() {
  asm volatile("s_wait_asynccnt 0x4" ::: "memory");  // older 4 copies landed
}
static __device__ __forceinline__ void wait_async0() {
  asm volatile("s_wait_asynccnt 0x0" ::: "memory");
}

// ---------------------------------------------------------------- convert
__global__ __launch_bounds__(256) void f32_to_bf16_kernel(
    const float* __restrict__ src, __bf16* __restrict__ dst, int n) {
  int i = blockIdx.x * blockDim.x + threadIdx.x;
  if (i < n) dst[i] = f2bf(src[i]);
}

// ---------------------------------------------------------------- GEMM
// C[M,N] = A[M,K] * B[N,K]^T ; A,B bf16 row-major (K contiguous).
// Block tile 128x128, 8 waves (4x2), each wave 32x64 (2x4 WMMA tiles).
// 2-stage LDS double buffer: async DMA of stage k+1 overlaps WMMA of stage k.
__global__ __launch_bounds__(256) void gemm_bf16_kernel(
    const __bf16* __restrict__ A, const __bf16* __restrict__ B,
    float* __restrict__ Cf, __bf16* __restrict__ Cb,
    int M, int N, int K)
{
  __shared__ __align__(64) __bf16 sA[2][128][32];   // 16 KB
  __shared__ __align__(64) __bf16 sB[2][128][32];   // 16 KB
  const int bm   = blockIdx.x * 128;
  const int bn   = blockIdx.y * 128;
  const int t    = threadIdx.x;
  const int lane = t & 31;
  const int wv   = t >> 5;
  const int wm   = (wv >> 1) * 32;     // 4 row groups of 32
  const int wn   = (wv & 1) * 64;      // 2 col groups of 64
  const int half = lane >> 4;
  const int l16  = lane & 15;

  v8f acc[2][4] = {};

  const int grow = t >> 1, gcol = (t & 1) * 16;  // 256 thr * 16 elem = 128x32
  const __bf16* ga = A + (size_t)(bm + grow) * K + gcol;
  const __bf16* gb = B + (size_t)(bn + grow) * K + gcol;

  // prologue: stage 0 in flight
  async_copy_b128(ga,     &sA[0][grow][gcol]);
  async_copy_b128(ga + 8, &sA[0][grow][gcol + 8]);
  async_copy_b128(gb,     &sB[0][grow][gcol]);
  async_copy_b128(gb + 8, &sB[0][grow][gcol + 8]);

  int stg = 0;
  for (int k0 = 0; k0 < K; k0 += 32, stg ^= 1) {
    if (k0 + 32 < K) {
      const __bf16* na = ga + k0 + 32;
      const __bf16* nb = gb + k0 + 32;
      async_copy_b128(na,     &sA[stg ^ 1][grow][gcol]);
      async_copy_b128(na + 8, &sA[stg ^ 1][grow][gcol + 8]);
      async_copy_b128(nb,     &sB[stg ^ 1][grow][gcol]);
      async_copy_b128(nb + 8, &sB[stg ^ 1][grow][gcol + 8]);
      __builtin_prefetch(na + 32, 0, 1);            // global_prefetch_b8
      __builtin_prefetch(nb + 32, 0, 1);
      wait_async_le4();   // in-order: stage `stg` complete, next 4 may fly
    } else {
      wait_async0();
    }
    __syncthreads();      // everyone's stage-`stg` copies visible

    v16bf af[2], bfr[4];
#pragma unroll
    for (int i = 0; i < 2; ++i)
      af[i] = load_frag2(&sA[stg][wm + i * 16 + l16][half * 8], 16);
#pragma unroll
    for (int j = 0; j < 4; ++j)
      bfr[j] = load_frag2(&sB[stg][wn + j * 16 + l16][half * 8], 16);
#pragma unroll
    for (int i = 0; i < 2; ++i)
#pragma unroll
      for (int j = 0; j < 4; ++j)
        acc[i][j] = wmma_bf16(af[i], bfr[j], acc[i][j]);
    __syncthreads();      // buffer `stg` free for reuse at k0+64
  }

#pragma unroll
  for (int i = 0; i < 2; ++i)
#pragma unroll
    for (int j = 0; j < 4; ++j) {
      const int rbase = bm + wm + i * 16 + half * 8;
      const int cbn   = bn + wn + j * 16 + l16;
#pragma unroll
      for (int r = 0; r < 8; ++r) {
        float v = acc[i][j][r];
        size_t idx = (size_t)(rbase + r) * N + cbn;
        if (Cf) Cf[idx] = v; else Cb[idx] = f2bf(v);
      }
    }
}

// ------------------------------------------------- RoPE + head-permute
// in: [s][h*64+d] bf16  ->  out: [h][s][64] bf16 (interleaved-pair rope)
__global__ __launch_bounds__(256) void rope_permute_kernel(
    const __bf16* __restrict__ in, __bf16* __restrict__ out,
    const int* __restrict__ pos)
{
  int idx = blockIdx.x * blockDim.x + threadIdx.x;
  if (idx >= S_LEN * NHEADS * (HDIM / 2)) return;
  int d2 = idx % (HDIM / 2);
  int h  = (idx / (HDIM / 2)) % NHEADS;
  int s  = idx / ((HDIM / 2) * NHEADS);
  float p  = (float)pos[s];
  float fr = __powf(10000.0f, -(float)(2 * d2) / (float)HDIM);
  float ang = p * fr;
  float c = __cosf(ang), sn = __sinf(ang);
  const __bf16* ip = in + (size_t)s * DMODEL + h * HDIM + 2 * d2;
  float e = bf2f(ip[0]), o = bf2f(ip[1]);
  __bf16* op = out + ((size_t)h * S_LEN + s) * HDIM + 2 * d2;
  op[0] = f2bf(e * c - o * sn);
  op[1] = f2bf(e * sn + o * c);
}

// ------------------------------------------------- V transpose
// in: [s][h*64+d] bf16 -> out: [h][64][s] bf16 (keys contiguous per dim)
__global__ __launch_bounds__(256) void vtrans_kernel(
    const __bf16* __restrict__ in, __bf16* __restrict__ out) {
  int idx = blockIdx.x * blockDim.x + threadIdx.x;
  if (idx >= S_LEN * DMODEL) return;
  int d = idx % HDIM;
  int h = (idx / HDIM) % NHEADS;
  int s = idx / DMODEL;
  out[((size_t)h * HDIM + d) * S_LEN + s] = in[idx];
}

// ------------------------------------------------- flash attention
// 1 wave = 16 query rows of one head; block = 4 waves (64 rows).
__global__ __launch_bounds__(128) void flash_attn_kernel(
    const __bf16* __restrict__ Qh, const __bf16* __restrict__ Kh,
    const __bf16* __restrict__ Vt, __bf16* __restrict__ attnO)
{
  __shared__ __align__(64) __bf16 Pls[4][16 * 32];   // per-wave P tile
  const int h    = blockIdx.y;
  const int wv   = threadIdx.x >> 5;
  const int lane = threadIdx.x & 31;
  const int half = lane >> 4, l16 = lane & 15;
  const int qbase = blockIdx.x * 64 + wv * 16;
  __bf16* P = Pls[wv];

  // Q A-fragments (16 rows x 64 K) -> 2 frags, loaded once
  const __bf16* qrow = Qh + ((size_t)h * S_LEN + qbase + l16) * HDIM;
  v16bf qf[2];
#pragma unroll
  for (int kk = 0; kk < 2; ++kk)
    qf[kk] = load_frag2(qrow + kk * 32 + half * 8, 16);

  v8f O[4] = {};
  float mrow[8], lrow[8];
#pragma unroll
  for (int r = 0; r < 8; ++r) { mrow[r] = -1e30f; lrow[r] = 0.f; }

  const int nchunks = (qbase + 15) / 32 + 1;   // causal bound
  for (int kb = 0; kb < nchunks; ++kb) {
    const int kbase = kb * 32;
    if (kb + 1 < nchunks) {     // warm next K/V chunk (global_prefetch_b8)
      __builtin_prefetch(Kh + ((size_t)h * S_LEN + kbase + 32 + l16) * HDIM, 0, 1);
      __builtin_prefetch(Vt + ((size_t)h * HDIM + l16) * S_LEN + kbase + 32, 0, 1);
    }
    v8f sacc[2] = {};
#pragma unroll
    for (int j = 0; j < 2; ++j) {
      const __bf16* kp = Kh + ((size_t)h * S_LEN + kbase + j * 16 + l16) * HDIM;
#pragma unroll
      for (int kk = 0; kk < 2; ++kk) {
        v16bf kf = load_frag2(kp + kk * 32 + half * 8, 16);
        sacc[j] = wmma_bf16(qf[kk], kf, sacc[j]);
      }
    }
    // online softmax (row = r + 8*half, cols spread over 16 lanes)
#pragma unroll
    for (int r = 0; r < 8; ++r) {
      const int m = qbase + r + 8 * half;
      float v0 = sacc[0][r] * 0.125f;             // 1/sqrt(64)
      float v1 = sacc[1][r] * 0.125f;
      if (kbase + l16 > m)      v0 = -1e30f;
      if (kbase + 16 + l16 > m) v1 = -1e30f;
      float rm = fmaxf(v0, v1);
#pragma unroll
      for (int d = 1; d < 16; d <<= 1) rm = fmaxf(rm, __shfl_xor(rm, d));
      float mnew  = fmaxf(mrow[r], rm);
      float alpha = __expf(mrow[r] - mnew);
      float p0 = __expf(v0 - mnew);
      float p1 = __expf(v1 - mnew);
      float rs = p0 + p1;
#pragma unroll
      for (int d = 1; d < 16; d <<= 1) rs += __shfl_xor(rs, d);
      lrow[r] = lrow[r] * alpha + rs;
      mrow[r] = mnew;
#pragma unroll
      for (int dt = 0; dt < 4; ++dt) O[dt][r] *= alpha;
      P[(r + 8 * half) * 32 + l16]      = f2bf(p0);
      P[(r + 8 * half) * 32 + 16 + l16] = f2bf(p1);
    }
    asm volatile("" ::: "memory");   // order per-wave LDS write -> read
    // P(16x32) @ V(32x64): A-frag from LDS, B-frags from Vt (contiguous keys)
    v16bf pf = load_frag2(&P[l16 * 32 + half * 8], 16);
#pragma unroll
    for (int dt = 0; dt < 4; ++dt) {
      const __bf16* vp = Vt + ((size_t)h * HDIM + dt * 16 + l16) * S_LEN
                       + kbase + half * 8;
      v16bf vf = load_frag2(vp, 16);
      O[dt] = wmma_bf16(pf, vf, O[dt]);
    }
    asm volatile("" ::: "memory");   // keep P alive until wmma consumed it
  }

#pragma unroll
  for (int r = 0; r < 8; ++r) {
    float inv = 1.0f / lrow[r];
    const int row = qbase + r + 8 * half;
#pragma unroll
    for (int dt = 0; dt < 4; ++dt)
      attnO[(size_t)row * DMODEL + h * HDIM + dt * 16 + l16] = f2bf(O[dt][r] * inv);
  }
}

// ---------------------------------------------------------------- launch
extern "C" void kernel_launch(void* const* d_in, const int* in_sizes, int n_in,
                              void* d_out, int out_size, void* d_ws, size_t ws_size,
                              hipStream_t stream) {
  (void)in_sizes; (void)n_in; (void)out_size; (void)ws_size;
  const float* x   = (const float*)d_in[0];
  const float* w_q = (const float*)d_in[1];
  const float* w_k = (const float*)d_in[2];
  const float* w_v = (const float*)d_in[3];
  const float* w_o = (const float*)d_in[4];
  const int*   tp  = (const int*)d_in[5];

  char* ws = (char*)d_ws;
  const size_t MB = 1u << 20;
  __bf16* xb  = (__bf16*)(ws + 0 * MB);   // 8 MB
  __bf16* wqb = (__bf16*)(ws + 8 * MB);   // 2 MB each
  __bf16* wkb = (__bf16*)(ws + 10 * MB);
  __bf16* wvb = (__bf16*)(ws + 12 * MB);
  __bf16* wob = (__bf16*)(ws + 14 * MB);
  __bf16* Qb  = (__bf16*)(ws + 16 * MB);  // 8 MB each
  __bf16* Kb  = (__bf16*)(ws + 24 * MB);
  __bf16* Vb  = (__bf16*)(ws + 32 * MB);
  __bf16* Qh  = (__bf16*)(ws + 40 * MB);
  __bf16* Kh  = (__bf16*)(ws + 48 * MB);
  __bf16* Vt  = (__bf16*)(ws + 56 * MB);
  __bf16* attn = Qb;                      // Qb is dead after rope -> reuse

  const int nx = S_LEN * DMODEL;          // 4,194,304
  const int nw = DMODEL * DMODEL;         // 1,048,576
  f32_to_bf16_kernel<<<(nx + 255) / 256, 256, 0, stream>>>(x,   xb,  nx);
  f32_to_bf16_kernel<<<(nw + 255) / 256, 256, 0, stream>>>(w_q, wqb, nw);
  f32_to_bf16_kernel<<<(nw + 255) / 256, 256, 0, stream>>>(w_k, wkb, nw);
  f32_to_bf16_kernel<<<(nw + 255) / 256, 256, 0, stream>>>(w_v, wvb, nw);
  f32_to_bf16_kernel<<<(nw + 255) / 256, 256, 0, stream>>>(w_o, wob, nw);

  dim3 ggrid(S_LEN / 128, DMODEL / 128);
  gemm_bf16_kernel<<<ggrid, 256, 0, stream>>>(xb, wqb, nullptr, Qb,
                                              S_LEN, DMODEL, DMODEL);
  gemm_bf16_kernel<<<ggrid, 256, 0, stream>>>(xb, wkb, nullptr, Kb,
                                              S_LEN, DMODEL, DMODEL);
  gemm_bf16_kernel<<<ggrid, 256, 0, stream>>>(xb, wvb, nullptr, Vb,
                                              S_LEN, DMODEL, DMODEL);

  const int nr = S_LEN * NHEADS * (HDIM / 2);
  rope_permute_kernel<<<(nr + 255) / 256, 256, 0, stream>>>(Qb, Qh, tp);
  rope_permute_kernel<<<(nr + 255) / 256, 256, 0, stream>>>(Kb, Kh, tp);
  vtrans_kernel<<<(nx + 255) / 256, 256, 0, stream>>>(Vb, Vt);

  flash_attn_kernel<<<dim3(S_LEN / 64, NHEADS), 128, 0, stream>>>(Qh, Kh, Vt, attn);

  gemm_bf16_kernel<<<ggrid, 256, 0, stream>>>(attn, wob, (float*)d_out, nullptr,
                                              S_LEN, DMODEL, DMODEL);
}
